// TxtNet_V2_88364657148581
// MI455X (gfx1250) — matrix operand order repaired and checked by reading, not compiled, not run on hardware
//
#include <hip/hip_runtime.h>
#include <hip/hip_bf16.h>

// ---------------- problem constants (from reference) ----------------
constexpr int NN    = 512;          // nodes
constexpr int EE    = 512;          // hyperedges
constexpr int KK    = 8;            // edges per node
constexpr int NNZ   = NN * KK;      // 4096 incidence pairs
constexpr int FTXT  = 1386;
constexpr int FPAD  = 1408;         // FTXT zero-padded to a multiple of 32
constexpr int HID   = 4096;
constexpr int HEADS = 8;
constexpr int CODE  = 64;
constexpr int WAOUT = HEADS * HID;  // 32768

// ---------------- CDNA5 WMMA types ----------------
typedef __attribute__((ext_vector_type(16))) __bf16 v16bf;
typedef __attribute__((ext_vector_type(8)))  float  v8f;

union BF16x16 { v16bf v; unsigned int u32[8]; unsigned short u[16]; };

// pack two f32 into one DWORD of bf16 pair (lo -> [15:0], hi -> [31:16]).
// One v_perm_b32 grabs the two high halves; +0x8000 first = round-half-up.
static __device__ __forceinline__ unsigned int pack2_bf16(float lo, float hi) {
    unsigned int ulo = __float_as_uint(lo) + 0x8000u;
    unsigned int uhi = __float_as_uint(hi) + 0x8000u;
    // byte table = {src0(uhi) bytes 7..4, src1(ulo) bytes 3..0}
    // result: byte0=ulo.b2, byte1=ulo.b3, byte2=uhi.b2, byte3=uhi.b3
    return __builtin_amdgcn_perm(uhi, ulo, 0x07060302u);
}

// ---------------------------------------------------------------
// bf16 WMMA GEMM, fp32 accumulate: C[M,N] = A[M,K]*B[K,N], all f32 in
// memory, cvt to bf16 in-register (v_perm DWORD pair packs). All dims are
// compile-time so strides fold into VMEM immediate offsets. KD must be a
// multiple of 32 (callers zero-pad) -> the kernel has zero predication.
// One wave computes a 16x64 C tile: one A fragment feeds 4 WMMAs.
// blockIdx.x = M tile (fast) so all M tiles of a B column strip reuse it
// from L2 after a single HBM read.
// ---------------------------------------------------------------
template<int M, int N, int KD, int LDA, int LDB, int LDC>
__global__ __launch_bounds__(32) void wmma_gemm(
    const float* __restrict__ A,
    const float* __restrict__ B,
    float* __restrict__ C)
{
    static_assert(KD % 32 == 0, "K must be padded to a multiple of 32");
    constexpr int NT = 64;                 // N tile per wave

    const int tileM = blockIdx.x * 16;
    const int tileN = blockIdx.y * NT;
    const int lane  = threadIdx.x & 31;
    const int half  = lane >> 4;           // 0: lanes 0-15, 1: lanes 16-31
    const int l16   = lane & 15;

    const float* __restrict__ Arow  = A + (size_t)(tileM + l16) * LDA;
    const float* __restrict__ Bbase = B + tileN + l16;

    v8f acc[4] = {v8f{}, v8f{}, v8f{}, v8f{}};

    for (int k0 = 0; k0 < KD; k0 += 32) {
        // ---- A fragment (contiguous K order, same as the B fragments) ----
        // lane<16 : u[0..15]=K0..15
        // lane>=16: u[0..15]=K16..31
        const int ka = k0 + half * 16;
        float av[16];
        if constexpr (LDA % 4 == 0) {
            *(float4*)(av + 0)  = *(const float4*)(Arow + ka);
            *(float4*)(av + 4)  = *(const float4*)(Arow + ka + 4);
            *(float4*)(av + 8)  = *(const float4*)(Arow + ka + 8);
            *(float4*)(av + 12) = *(const float4*)(Arow + ka + 12);
        } else {                 // LDA even -> rows 8B aligned -> b64 loads
#pragma unroll
            for (int t = 0; t < 4; ++t) {
                *(float2*)(av + 2 * t)     = *(const float2*)(Arow + ka + 2 * t);
                *(float2*)(av + 8 + 2 * t) = *(const float2*)(Arow + ka + 8 + 2 * t);
            }
        }
        BF16x16 a;
#pragma unroll
        for (int j = 0; j < 8; ++j)
            a.u32[j] = pack2_bf16(av[2 * j], av[2 * j + 1]);

        // ---- B fragments, 4 sub-tiles of 16 cols ----
        // lane<16 : col, K=k0+0..15 ; lane>=16: col, K=k0+16..31
        const int kb = k0 + half * 16;
        const float* __restrict__ Bp = Bbase + (size_t)kb * LDB;
        BF16x16 b[4];
#pragma unroll
        for (int c = 0; c < 4; ++c)
#pragma unroll
            for (int j = 0; j < 8; ++j)
                b[c].u32[j] = pack2_bf16(Bp[(size_t)(2 * j) * LDB + 16 * c],
                                         Bp[(size_t)(2 * j + 1) * LDB + 16 * c]);

        if (k0 + 32 < KD)        // prefetch next B slab (global_prefetch_b8)
            __builtin_prefetch(Bp + (size_t)32 * LDB, 0, 1);

#pragma unroll
        for (int c = 0; c < 4; ++c)
            acc[c] = __builtin_amdgcn_wmma_f32_16x16x32_bf16(
                         false, a.v, false, b[c].v, (short)0, acc[c], false, false);
    }

    // C layout: VGPR j -> row tileM + j + 8*half, col tileN + l16 + 16*c
#pragma unroll
    for (int j = 0; j < 8; ++j) {
        float* Crow = C + (size_t)(tileM + j + 8 * half) * LDC + tileN + l16;
#pragma unroll
        for (int c = 0; c < 4; ++c) Crow[16 * c] = acc[c][j];
    }
}

// ---------------- K padding helpers (remove GEMM tail entirely) ----------------
// pad columns: dst[r, c] = c < cols_src ? src[r, c] : 0   (dst is rows x cols_pad)
__global__ void pad_cols(const float* __restrict__ src, float* __restrict__ dst,
                         int rows, int cols_src, int cols_pad)
{
    size_t t = (size_t)blockIdx.x * blockDim.x + threadIdx.x;
    size_t total = (size_t)rows * cols_pad;
    if (t >= total) return;
    int r = (int)(t / cols_pad), c = (int)(t % cols_pad);
    dst[t] = (c < cols_src) ? src[(size_t)r * cols_src + c] : 0.0f;
}

// pad rows: dst[r, c] = r < rows_src ? src[r, c] : 0   (dst is rows_pad x cols)
__global__ void pad_rows(const float* __restrict__ src, float* __restrict__ dst,
                         int rows_src, int rows_pad, int cols)
{
    size_t t = (size_t)blockIdx.x * blockDim.x + threadIdx.x;
    size_t total = (size_t)rows_pad * cols;
    if (t >= total) return;
    int r = (int)(t / cols);
    dst[t] = (r < rows_src) ? src[t] : 0.0f;
}

// ---------------- graph norm helpers ----------------
__global__ void degree_kernel(const int* __restrict__ node_idx,
                              const int* __restrict__ edge_idx,
                              int* dcnt, int* bcnt, int nnz)
{
    int i = blockIdx.x * blockDim.x + threadIdx.x;
    if (i < nnz) {
        atomicAdd(&dcnt[node_idx[i]], 1);
        atomicAdd(&bcnt[edge_idx[i]], 1);
    }
}

__global__ void norms_kernel(const int* dcnt, const int* bcnt,
                             float* Dinv, float* Binv, int n)
{
    int i = blockIdx.x * blockDim.x + threadIdx.x;
    if (i < n) {
        int d = dcnt[i], b = bcnt[i];
        Dinv[i] = d > 0 ? 1.0f / (float)d : 0.0f;
        Binv[i] = b > 0 ? 1.0f / (float)b : 0.0f;
    }
}

// scatter: dst[edge_idx[i], :] += src[node_idx[i], :]
__global__ void scatter_add_rows(const float* __restrict__ src,
                                 const int* __restrict__ node_idx,
                                 const int* __restrict__ edge_idx,
                                 float* __restrict__ dst, int ncols)
{
    int i = blockIdx.x;
    int n = node_idx[i], e = edge_idx[i];
    const float* s = src + (size_t)n * ncols;
    float*       d = dst + (size_t)e * ncols;
    for (int c = threadIdx.x; c < ncols; c += blockDim.x)
        atomicAdd(&d[c], s[c]);
}

__global__ void scale_rows(float* __restrict__ m, const float* __restrict__ scale,
                           int ncols)
{
    int e = blockIdx.x;
    float sc = scale[e];
    for (int c = threadIdx.x; c < ncols; c += blockDim.x)
        m[(size_t)e * ncols + c] *= sc;
}

// node gather of hgc1: feat = relu(Dinv * sum_j edge_feat[e_j]*Binv[e_j] + b1)
__global__ void gather_node_relu(const float* __restrict__ edge_feat,
                                 const int* __restrict__ edge_idx,
                                 const float* __restrict__ Binv,
                                 const float* __restrict__ Dinv,
                                 const float* __restrict__ b1,
                                 float* __restrict__ feat_out,
                                 float* __restrict__ acat_out)
{
    int n = blockIdx.x;
    __shared__ int   es[KK];
    __shared__ float bs[KK];
    if (threadIdx.x < KK) {
        int e = edge_idx[n * KK + threadIdx.x];
        es[threadIdx.x] = e;
        bs[threadIdx.x] = Binv[e];
    }
    __syncthreads();
    float dinv = Dinv[n];
    for (int d = threadIdx.x; d < HID; d += blockDim.x) {
        float s = 0.0f;
#pragma unroll
        for (int j = 0; j < KK; ++j)
            s += edge_feat[(size_t)es[j] * HID + d] * bs[j];
        float v = s * dinv + b1[d];
        v = v > 0.0f ? v : 0.0f;
        feat_out[(size_t)n * HID + d] = v;
        acat_out[(size_t)n * HID + d] = v;
    }
}

// sx[n,h] = dot(Y[n, h*HID:], att[h,:HID]); se[e,h] = dot(Y[NN+e,...], att[h,HID:])
__global__ void head_dot_kernel(const float* __restrict__ Y,
                                const float* __restrict__ att,
                                float* __restrict__ sx, float* __restrict__ se)
{
    int row = blockIdx.x;            // 0..2*NN-1
    int h   = blockIdx.y;
    bool is_edge = row >= NN;
    const float* y = Y + (size_t)row * WAOUT + (size_t)h * HID;
    const float* a = att + (size_t)h * (2 * HID) + (is_edge ? HID : 0);
    float s = 0.0f;
    for (int d = threadIdx.x; d < HID; d += blockDim.x) s += y[d] * a[d];
    __shared__ float red[256];
    red[threadIdx.x] = s;
    __syncthreads();
    for (int o = 128; o > 0; o >>= 1) {
        if (threadIdx.x < o) red[threadIdx.x] += red[threadIdx.x + o];
        __syncthreads();
    }
    if (threadIdx.x == 0) {
        if (is_edge) se[(size_t)(row - NN) * HEADS + h] = red[0];
        else         sx[(size_t)row * HEADS + h]        = red[0];
    }
}

// leaky_relu + segment softmax; node groups are contiguous runs of KK pairs
__global__ void attn_softmax_kernel(const float* __restrict__ sx,
                                    const float* __restrict__ se,
                                    const int* __restrict__ edge_idx,
                                    float* __restrict__ alpha)
{
    int t = blockIdx.x * blockDim.x + threadIdx.x;
    if (t >= NN * HEADS) return;
    int n = t / HEADS, h = t % HEADS;
    float lg[KK];
    float mx = -1e30f;
    float sxv = sx[(size_t)n * HEADS + h];
#pragma unroll
    for (int j = 0; j < KK; ++j) {
        int e = edge_idx[n * KK + j];
        float v = sxv + se[(size_t)e * HEADS + h];
        v = v > 0.0f ? v : 0.2f * v;     // leaky_relu(0.2)
        lg[j] = v;
        mx = fmaxf(mx, v);
    }
    float s = 0.0f;
#pragma unroll
    for (int j = 0; j < KK; ++j) { lg[j] = expf(lg[j] - mx); s += lg[j]; }
    float inv = 1.0f / s;
#pragma unroll
    for (int j = 0; j < KK; ++j)
        alpha[(size_t)(n * KK + j) * HEADS + h] = lg[j] * inv;
}

// edge_m[e,h,:] += alpha[i,h] * xh[node_idx[i], h, :]
__global__ void edge_msg_scatter(const float* __restrict__ Y,   // xh = rows 0..NN-1
                                 const float* __restrict__ alpha,
                                 const int* __restrict__ node_idx,
                                 const int* __restrict__ edge_idx,
                                 float* __restrict__ edge_m)
{
    int i = blockIdx.x, h = blockIdx.y;
    int n = node_idx[i], e = edge_idx[i];
    float a = alpha[(size_t)i * HEADS + h];
    const float* x = Y + (size_t)n * WAOUT + (size_t)h * HID;
    float* dst = edge_m + ((size_t)e * HEADS + h) * HID;
    for (int d = threadIdx.x; d < HID; d += blockDim.x)
        atomicAdd(&dst[d], a * x[d]);
}

// hid_cat = feat + mean_h( Dinv * sum_j alpha*edge_m[e_j]*Binv[e_j] ) + ba
__global__ void node_msg_gather(const float* __restrict__ edge_m,
                                const float* __restrict__ alpha,
                                const int* __restrict__ edge_idx,
                                const float* __restrict__ Binv,
                                const float* __restrict__ Dinv,
                                const float* __restrict__ feat,
                                const float* __restrict__ ba,
                                float* __restrict__ hid_cat)
{
    int n = blockIdx.x;
    __shared__ int   es[KK];
    __shared__ float bs[KK];
    __shared__ float al[KK * HEADS];
    if (threadIdx.x < KK) {
        int e = edge_idx[n * KK + threadIdx.x];
        es[threadIdx.x] = e;
        bs[threadIdx.x] = Binv[e];
    }
    if (threadIdx.x < KK * HEADS) {
        int j = threadIdx.x / HEADS, h = threadIdx.x % HEADS;
        al[threadIdx.x] = alpha[(size_t)(n * KK + j) * HEADS + h];
    }
    __syncthreads();
    float scale = Dinv[n] * (1.0f / (float)HEADS);
    for (int d = threadIdx.x; d < HID; d += blockDim.x) {
        float tot = 0.0f;
#pragma unroll
        for (int h = 0; h < HEADS; ++h) {
#pragma unroll
            for (int j = 0; j < KK; ++j)
                tot += al[j * HEADS + h] *
                       edge_m[((size_t)es[j] * HEADS + h) * HID + d] * bs[j];
        }
        hid_cat[(size_t)n * HID + d] =
            feat[(size_t)n * HID + d] + tot * scale + ba[d];
    }
}

// final node gather of hgc2 + bias + tanh
__global__ void final_out(const float* __restrict__ edge2,
                          const int* __restrict__ edge_idx,
                          const float* __restrict__ Binv,
                          const float* __restrict__ Dinv,
                          const float* __restrict__ b2,
                          float* __restrict__ hid_out,
                          float* __restrict__ code_out)
{
    int n = blockIdx.x;
    int c = threadIdx.x;               // blockDim = CODE
    float s = 0.0f;
#pragma unroll
    for (int j = 0; j < KK; ++j) {
        int e = edge_idx[n * KK + j];
        s += edge2[(size_t)e * CODE + c] * Binv[e];
    }
    float v = s * Dinv[n] + b2[c];
    hid_out[(size_t)n * CODE + c]  = v;
    code_out[(size_t)n * CODE + c] = tanhf(v);
}

// ---------------------------------------------------------------
extern "C" void kernel_launch(void* const* d_in, const int* in_sizes, int n_in,
                              void* d_out, int out_size, void* d_ws, size_t ws_size,
                              hipStream_t stream) {
    (void)in_sizes; (void)n_in; (void)out_size; (void)ws_size;

    const float* x        = (const float*)d_in[0];
    const int*   node_idx = (const int*)  d_in[1];
    const int*   edge_idx = (const int*)  d_in[2];
    const float* W1       = (const float*)d_in[3];
    const float* b1       = (const float*)d_in[4];
    const float* Wa       = (const float*)d_in[5];
    const float* att      = (const float*)d_in[6];
    const float* ba       = (const float*)d_in[7];
    const float* W2       = (const float*)d_in[8];
    const float* b2       = (const float*)d_in[9];

    float* out_feat = (float*)d_out;                       // [NN, HID]
    float* out_hid  = out_feat + (size_t)NN * HID;         // [NN, CODE]
    float* out_code = out_hid  + (size_t)NN * CODE;        // [NN, CODE]

    char* ws = (char*)d_ws;
    size_t off = 0;
    auto alloc = [&](size_t bytes) -> void* {
        void* p = ws + off;
        off = (off + bytes + 255) & ~(size_t)255;
        return p;
    };
    float* xpad      = (float*)alloc((size_t)NN * FPAD * 4);       // x, K padded
    float* w1pad     = (float*)alloc((size_t)FPAD * HID * 4);      // W1, K padded
    float* xl        = (float*)alloc((size_t)NN * HID * 4);        // x @ W1
    float* Acat      = (float*)alloc((size_t)2 * NN * HID * 4);    // [feat; he_attr]
    float* edge_feat = (float*)alloc((size_t)EE * HID * 4);
    float* Y         = (float*)alloc((size_t)2 * NN * WAOUT * 4);  // [xh; eh]
    float* edge_m    = (float*)alloc((size_t)EE * HEADS * HID * 4);
    float* hid_cat   = (float*)alloc((size_t)NN * HID * 4);
    float* t2        = (float*)alloc((size_t)NN * CODE * 4);       // hid_cat @ W2
    float* edge2     = (float*)alloc((size_t)EE * CODE * 4);
    float* sx        = (float*)alloc((size_t)NN * HEADS * 4);
    float* se        = (float*)alloc((size_t)EE * HEADS * 4);
    float* alpha     = (float*)alloc((size_t)NNZ * HEADS * 4);
    int*   dcnt      = (int*)  alloc((size_t)NN * 4);
    int*   bcnt      = (int*)  alloc((size_t)EE * 4);
    float* Dinv      = (float*)alloc((size_t)NN * 4);
    float* Binv      = (float*)alloc((size_t)EE * 4);
    float* he_attr   = Acat + (size_t)NN * HID;

    // zero accumulation buffers (graph-capture-safe)
    hipMemsetAsync(dcnt, 0, (size_t)NN * 4, stream);
    hipMemsetAsync(bcnt, 0, (size_t)EE * 4, stream);
    hipMemsetAsync(edge_feat, 0, (size_t)EE * HID * 4, stream);
    hipMemsetAsync(he_attr,   0, (size_t)NN * HID * 4, stream);
    hipMemsetAsync(edge_m,    0, (size_t)EE * HEADS * HID * 4, stream);
    hipMemsetAsync(edge2,     0, (size_t)EE * CODE * 4, stream);

    // norms
    degree_kernel<<<(NNZ + 255) / 256, 256, 0, stream>>>(node_idx, edge_idx, dcnt, bcnt, NNZ);
    norms_kernel<<<(NN + 255) / 256, 256, 0, stream>>>(dcnt, bcnt, Dinv, Binv, NN);

    // K-pad inputs of GEMM1 so the WMMA loop needs no tail
    {
        size_t tx = (size_t)NN * FPAD;
        pad_cols<<<(int)((tx + 255) / 256), 256, 0, stream>>>(x, xpad, NN, FTXT, FPAD);
        size_t tw = (size_t)FPAD * HID;
        pad_rows<<<(int)((tw + 255) / 256), 256, 0, stream>>>(W1, w1pad, FTXT, FPAD, HID);
    }

    // hgc1: xl = x @ W1 ; edge scatter ; node gather + relu -> feat
    wmma_gemm<NN, HID, FPAD, FPAD, HID, HID>
        <<<dim3(NN / 16, HID / 64), 32, 0, stream>>>(xpad, w1pad, xl);
    scatter_add_rows<<<NNZ, 256, 0, stream>>>(xl, node_idx, edge_idx, edge_feat, HID);
    gather_node_relu<<<NN, 256, 0, stream>>>(edge_feat, edge_idx, Binv, Dinv, b1,
                                             out_feat, Acat);

    // he_attr = segment-mean of feat by edge (1/ecnt == Binv)
    scatter_add_rows<<<NNZ, 256, 0, stream>>>(Acat, node_idx, edge_idx, he_attr, HID);
    scale_rows<<<EE, 256, 0, stream>>>(he_attr, Binv, HID);

    // big fused GEMM: Y = [feat; he_attr] @ Wa  (Wa streamed from HBM once)
    wmma_gemm<2 * NN, WAOUT, HID, HID, WAOUT, WAOUT>
        <<<dim3(2 * NN / 16, WAOUT / 64), 32, 0, stream>>>(Acat, Wa, Y);

    // attention
    head_dot_kernel<<<dim3(2 * NN, HEADS), 256, 0, stream>>>(Y, att, sx, se);
    attn_softmax_kernel<<<(NN * HEADS + 255) / 256, 256, 0, stream>>>(sx, se, edge_idx, alpha);
    edge_msg_scatter<<<dim3(NNZ, HEADS), 256, 0, stream>>>(Y, alpha, node_idx, edge_idx, edge_m);
    node_msg_gather<<<NN, 256, 0, stream>>>(edge_m, alpha, edge_idx, Binv, Dinv,
                                            Acat, ba, hid_cat);

    // hgc2: t2 = hid_cat @ W2 ; edge scatter ; final gather + tanh
    wmma_gemm<NN, CODE, HID, HID, CODE, CODE>
        <<<dim3(NN / 16, CODE / 64), 32, 0, stream>>>(hid_cat, W2, t2);
    scatter_add_rows<<<NNZ, 64, 0, stream>>>(t2, node_idx, edge_idx, edge2, CODE);
    final_out<<<NN, CODE, 0, stream>>>(edge2, edge_idx, Binv, Dinv, b2, out_hid, out_code);
}
